// MOMEBlock_74689481277466
// MI455X (gfx1250) — compile-verified, matrix operand order checked by reading, not run on hardware
//
#include <hip/hip_runtime.h>
#include <hip/hip_bf16.h>

typedef __attribute__((ext_vector_type(16))) __bf16 bf16x16;
typedef __attribute__((ext_vector_type(8)))  __bf16 bf16x8;
typedef __attribute__((ext_vector_type(8)))  float  f32x8;

static __device__ inline bf16x16 load_frag(const __bf16* p0, const __bf16* p1) {
    bf16x8 lo = *(const bf16x8*)p0;
    bf16x8 hi = *(const bf16x8*)p1;
    bf16x16 r;
#pragma unroll
    for (int i = 0; i < 8; ++i) { r[i] = lo[i]; r[i + 8] = hi[i]; }
    return r;
}

static __device__ inline f32x8 wmma_bf16(bf16x16 a, bf16x16 b, f32x8 c) {
    return __builtin_amdgcn_wmma_f32_16x16x32_bf16(false, a, false, b, (short)0, c, false, false);
}

// ---- DPP16 butterfly reductions over each 16-lane group (VALU, no LDS) ----
// quad_perm(1,0,3,2)=0xB1, quad_perm(2,3,0,1)=0x4E, row_half_mirror=0x141, row_mirror=0x140
template <int CTRL>
static __device__ inline float dpp_xchg(float x) {
    int i = __builtin_amdgcn_update_dpp(0, __float_as_int(x), CTRL, 0xf, 0xf, true);
    return __int_as_float(i);
}
static __device__ inline float rowmax16(float x) {
    x = fmaxf(x, dpp_xchg<0xB1>(x));
    x = fmaxf(x, dpp_xchg<0x4E>(x));
    x = fmaxf(x, dpp_xchg<0x141>(x));
    x = fmaxf(x, dpp_xchg<0x140>(x));
    return x;
}
static __device__ inline float rowsum16(float x) {
    x += dpp_xchg<0xB1>(x);
    x += dpp_xchg<0x4E>(x);
    x += dpp_xchg<0x141>(x);
    x += dpp_xchg<0x140>(x);
    return x;
}

// ---------------------------------------------------------------- weight cvt
__global__ void f32_to_bf16_kernel(const float* __restrict__ in, __bf16* __restrict__ out, int n) {
    int i = blockIdx.x * 256 + threadIdx.x;
    if (i < n) out[i] = (__bf16)in[i];
}

// ---------------------------------------------------------------- layernorm
__global__ __launch_bounds__(256)
void layernorm_bf16_kernel(const float* __restrict__ x, const float* __restrict__ g,
                           const float* __restrict__ b, __bf16* __restrict__ out, int C) {
    const int row = blockIdx.x;
    const float* xr = x + (size_t)row * C;
    float sum = 0.f, sq = 0.f;
    for (int i = threadIdx.x; i < C; i += 256) { float v = xr[i]; sum += v; sq += v * v; }
#pragma unroll
    for (int off = 16; off; off >>= 1) {
        sum += __shfl_xor(sum, off, 32);
        sq  += __shfl_xor(sq,  off, 32);
    }
    __shared__ float ssum[8], ssq[8];
    if ((threadIdx.x & 31) == 0) { ssum[threadIdx.x >> 5] = sum; ssq[threadIdx.x >> 5] = sq; }
    __syncthreads();
    sum = 0.f; sq = 0.f;
#pragma unroll
    for (int i = 0; i < 8; ++i) { sum += ssum[i]; sq += ssq[i]; }
    const float mu  = sum / C;
    const float var = sq / C - mu * mu;
    const float inv = rsqrtf(var + 1e-5f);
    __bf16* orow = out + (size_t)row * C;
    for (int i = threadIdx.x; i < C; i += 256)
        orow[i] = (__bf16)((xr[i] - mu) * inv * g[i] + b[i]);
}

// ---------------------------------------------------------------- GEMM (X @ W^T)
// 8 waves/block, tile = 128 rows x 64 cols, k-step 64, double-buffered W slab in LDS.
enum { MODE_QKV = 0, MODE_PROJ = 1, MODE_GELU = 2, MODE_OUT2 = 3 };

template <int MODE>
__global__ __launch_bounds__(256)
void gemm_bf16_kernel(const __bf16* __restrict__ A, const __bf16* __restrict__ W,
                      int M, int N, int K,
                      const float* __restrict__ bias, const float* __restrict__ resid,
                      float* __restrict__ fout, __bf16* __restrict__ bout,
                      __bf16* __restrict__ qs, __bf16* __restrict__ kmat,
                      __bf16* __restrict__ vT) {
    __shared__ __bf16 wtile[2][64 * 64];
    const int tid  = threadIdx.x;
    const int wave = tid >> 5, lane = tid & 31;
    const int half = lane >> 4, l15 = lane & 15;
    const int nBase = blockIdx.x * 64;
    const int mBase = blockIdx.y * 128;
    const __bf16* Arow = A + (size_t)(mBase + wave * 16 + l15) * K;

    f32x8 acc[4];
#pragma unroll
    for (int t = 0; t < 4; ++t)
#pragma unroll
        for (int r = 0; r < 8; ++r) acc[t][r] = 0.f;

    const int ldr = tid >> 2;          // 0..63 (W row within tile)
    const int ldc = (tid & 3) << 4;    // 0,16,32,48 (k offset, 16 bf16 per thread)
    const __bf16* wsrc = W + (size_t)(nBase + ldr) * K + ldc;

    // preload slab 0
    *(bf16x8*)&wtile[0][ldr * 64 + ldc]     = *(const bf16x8*)(wsrc);
    *(bf16x8*)&wtile[0][ldr * 64 + ldc + 8] = *(const bf16x8*)(wsrc + 8);
    __syncthreads();

    const int T = K >> 6;
    for (int kt = 0; kt < T; ++kt) {
        bf16x8 nx0, nx1;
        if (kt + 1 < T) {
            const __bf16* ws = wsrc + (size_t)(kt + 1) * 64;
            nx0 = *(const bf16x8*)(ws);
            nx1 = *(const bf16x8*)(ws + 8);
            if (kt + 2 < T) __builtin_prefetch(wsrc + (size_t)(kt + 2) * 64, 0, 1);
        }
        const __bf16* wt = &wtile[kt & 1][0];
#pragma unroll
        for (int sub = 0; sub < 2; ++sub) {
            const int kb = kt * 64 + sub * 32 + half * 8;
            const bf16x16 af = load_frag(Arow + kb, Arow + kb + 16);
            const int lb = sub * 32 + half * 8;
#pragma unroll
            for (int t = 0; t < 4; ++t) {
                const int wr = t * 16 + l15;
                const bf16x16 bf = load_frag(&wt[wr * 64 + lb], &wt[wr * 64 + lb + 16]);
                acc[t] = wmma_bf16(af, bf, acc[t]);
            }
        }
        if (kt + 1 < T) {
            __bf16* wd = &wtile[(kt + 1) & 1][ldr * 64 + ldc];
            *(bf16x8*)wd       = nx0;
            *(bf16x8*)(wd + 8) = nx1;
        }
        __syncthreads();
    }

#pragma unroll
    for (int t = 0; t < 4; ++t) {
        const int col = nBase + t * 16 + l15;
#pragma unroll
        for (int r = 0; r < 8; ++r) {
            const int row = mBase + wave * 16 + r + half * 8;
            float v = acc[t][r];
            if (MODE == MODE_QKV) {
                // col in [0,3072): 0..1023 = q, 1024..2047 = k, 2048..3071 = v
                const int which = col >> 10, c = col & 1023, hh = c >> 6, d = c & 63;
                const int b = row >> 10, nn = row & 1023;
                if (which == 0)
                    qs[(((size_t)b * 16 + hh) * 1024 + nn) * 64 + d] = (__bf16)(v * 0.125f);
                else if (which == 1)
                    kmat[(((size_t)b * 16 + hh) * 1024 + nn) * 64 + d] = (__bf16)v;
                else
                    vT[(((size_t)b * 16 + hh) * 64 + d) * 1024 + nn] = (__bf16)v;
            } else if (MODE == MODE_PROJ) {
                v += bias[col] + resid[(size_t)row * N + col];
                fout[(size_t)row * N + col] = v;
            } else if (MODE == MODE_GELU) {
                v += bias[col];
                v = 0.5f * v * (1.0f + erff(v * 0.70710678f));
                bout[(size_t)row * N + col] = (__bf16)v;
            } else { // MODE_OUT2: out = 2m, t = m
                v += bias[col];
                fout[(size_t)row * N + col] = v + v;
                fout[(size_t)4194304 + (size_t)row * N + col] = v;
            }
        }
    }
}

// ---------------------------------------------------------------- attention
// One wave per (b, h, 16-row n tile). Online softmax, m-step of 32 keys.
__global__ __launch_bounds__(256)
void attention_kernel(const __bf16* __restrict__ qs, const __bf16* __restrict__ kmat,
                      const __bf16* __restrict__ vT, const float* __restrict__ rpb,
                      const int* __restrict__ pmask, __bf16* __restrict__ aout) {
    __shared__ __bf16 pbuf[8][16 * 32];
    const int tid  = threadIdx.x;
    const int wave = tid >> 5, lane = tid & 31;
    const int half = lane >> 4, l15 = lane & 15;
    const int wid = blockIdx.x * 8 + wave;   // 0..4095
    const int b   = wid >> 10;
    const int rem = wid & 1023;
    const int h   = rem >> 6;
    const int n0  = (rem & 63) << 4;
    const size_t bh = (size_t)b * 16 + h;

    const __bf16* qrow = qs + (bh * 1024 + n0 + l15) * 64;
    const bf16x16 aq0 = load_frag(qrow + half * 8,      qrow + half * 8 + 16);
    const bf16x16 aq1 = load_frag(qrow + 32 + half * 8, qrow + 32 + half * 8 + 16);

    f32x8 o[4];
    float Mx[8], Sx[8];
#pragma unroll
    for (int t = 0; t < 4; ++t)
#pragma unroll
        for (int r = 0; r < 8; ++r) o[t][r] = 0.f;
#pragma unroll
    for (int r = 0; r < 8; ++r) { Mx[r] = -1e30f; Sx[r] = 0.f; }

    const float* biasbase = rpb + (size_t)h * 1024 * 1024;
    __bf16* pb = &pbuf[wave][0];

    for (int mt = 0; mt < 1024; mt += 32) {
        f32x8 s0, s1;
#pragma unroll
        for (int r = 0; r < 8; ++r) { s0[r] = 0.f; s1[r] = 0.f; }
        {
            const __bf16* k0p = kmat + (bh * 1024 + mt + l15) * 64;
            const __bf16* k1p = k0p + 16 * 64;
            if (mt + 32 < 1024) __builtin_prefetch(k0p + 32 * 64, 0, 1);
            bf16x16 bk;
            bk = load_frag(k0p + half * 8,      k0p + half * 8 + 16);      s0 = wmma_bf16(aq0, bk, s0);
            bk = load_frag(k0p + 32 + half * 8, k0p + 32 + half * 8 + 16); s0 = wmma_bf16(aq1, bk, s0);
            bk = load_frag(k1p + half * 8,      k1p + half * 8 + 16);      s1 = wmma_bf16(aq0, bk, s1);
            bk = load_frag(k1p + 32 + half * 8, k1p + 32 + half * 8 + 16); s1 = wmma_bf16(aq1, bk, s1);
        }
        const int c0 = mt + l15, c1 = mt + 16 + l15;
        const int pm0 = pmask[b * 1024 + c0];
        const int pm1 = pmask[b * 1024 + c1];
#pragma unroll
        for (int r = 0; r < 8; ++r) {
            const int row = n0 + r + half * 8;
            s0[r] = pm0 ? (s0[r] + biasbase[(size_t)row * 1024 + c0]) : -1e30f;
            s1[r] = pm1 ? (s1[r] + biasbase[(size_t)row * 1024 + c1]) : -1e30f;
        }
        float pr0[8], pr1[8];
#pragma unroll
        for (int r = 0; r < 8; ++r) {
            float m = rowmax16(fmaxf(s0[r], s1[r]));     // DPP butterfly, no LDS
            m = fmaxf(m, Mx[r]);
            const float sc = __expf(Mx[r] - m);
            const float p0 = __expf(s0[r] - m);
            const float p1 = __expf(s1[r] - m);
            const float rs = rowsum16(p0 + p1);          // DPP butterfly, no LDS
            Sx[r] = Sx[r] * sc + rs;
            Mx[r] = m;
            pr0[r] = p0; pr1[r] = p1;
#pragma unroll
            for (int t = 0; t < 4; ++t) o[t][r] *= sc;
        }
        // P tile (16x32) through per-wave LDS: D-layout -> A-fragment layout
#pragma unroll
        for (int r = 0; r < 8; ++r) {
            const int rr = r + half * 8;
            pb[rr * 32 + l15]      = (__bf16)pr0[r];
            pb[rr * 32 + 16 + l15] = (__bf16)pr1[r];
        }
        asm volatile("s_wait_dscnt 0" ::: "memory");
        const bf16x16 ap = load_frag(&pb[l15 * 32 + half * 8],
                                     &pb[l15 * 32 + half * 8 + 16]);
#pragma unroll
        for (int t = 0; t < 4; ++t) {
            const __bf16* vp = vT + (bh * 64 + t * 16 + l15) * 1024 + mt + half * 8;
            const bf16x16 bv = load_frag(vp, vp + 16);
            o[t] = wmma_bf16(ap, bv, o[t]);
        }
    }

#pragma unroll
    for (int t = 0; t < 4; ++t) {
        const int col = h * 64 + t * 16 + l15;
#pragma unroll
        for (int r = 0; r < 8; ++r) {
            const int row = n0 + r + half * 8;
            aout[((size_t)b * 1024 + row) * 1024 + col] = (__bf16)(o[t][r] / Sx[r]);
        }
    }
}

// ---------------------------------------------------------------- launch
extern "C" void kernel_launch(void* const* d_in, const int* in_sizes, int n_in,
                              void* d_out, int out_size, void* d_ws, size_t ws_size,
                              hipStream_t stream) {
    const float* x     = (const float*)d_in[0];
    const int*   pmask = (const int*)  d_in[1];
    const float* rpb   = (const float*)d_in[2];
    const float* w_qkv = (const float*)d_in[3];
    const float* w_proj= (const float*)d_in[4];
    const float* b_proj= (const float*)d_in[5];
    const float* ln1_g = (const float*)d_in[6];
    const float* ln1_b = (const float*)d_in[7];
    const float* ln2_g = (const float*)d_in[8];
    const float* ln2_b = (const float*)d_in[9];
    const float* w_fc1 = (const float*)d_in[10];
    const float* b_fc1 = (const float*)d_in[11];
    const float* w_fc2 = (const float*)d_in[12];
    const float* b_fc2 = (const float*)d_in[13];
    float* out = (float*)d_out;

    const int B = 4, N = 1024, C = 1024, H = 16, HID = 4096;
    const int M = B * N; // 4096 rows

    char* p = (char*)d_ws;
    __bf16* h1    = (__bf16*)p; p += (size_t)M * C * 2;        // 8 MB
    __bf16* qs    = (__bf16*)p; p += (size_t)B * H * N * 64 * 2;
    __bf16* kbuf  = (__bf16*)p; p += (size_t)B * H * N * 64 * 2;
    __bf16* vT    = (__bf16*)p; p += (size_t)B * H * 64 * N * 2;
    __bf16* abuf  = (__bf16*)p; p += (size_t)M * C * 2;
    float*  x2    = (float*) p; p += (size_t)M * C * 4;        // 16 MB
    __bf16* h2    = (__bf16*)p; p += (size_t)M * C * 2;
    __bf16* u     = (__bf16*)p; p += (size_t)M * HID * 2;      // 32 MB
    __bf16* wqkvb = (__bf16*)p; p += (size_t)3 * C * C * 2;
    __bf16* wprjb = (__bf16*)p; p += (size_t)C * C * 2;
    __bf16* wfc1b = (__bf16*)p; p += (size_t)HID * C * 2;
    __bf16* wfc2b = (__bf16*)p; p += (size_t)C * HID * 2;

    // weights -> bf16
    f32_to_bf16_kernel<<<(3 * C * C + 255) / 256, 256, 0, stream>>>(w_qkv,  wqkvb, 3 * C * C);
    f32_to_bf16_kernel<<<(C * C     + 255) / 256, 256, 0, stream>>>(w_proj, wprjb, C * C);
    f32_to_bf16_kernel<<<(HID * C   + 255) / 256, 256, 0, stream>>>(w_fc1,  wfc1b, HID * C);
    f32_to_bf16_kernel<<<(C * HID   + 255) / 256, 256, 0, stream>>>(w_fc2,  wfc2b, C * HID);

    // LN1
    layernorm_bf16_kernel<<<M, 256, 0, stream>>>(x, ln1_g, ln1_b, h1, C);

    // QKV GEMM: [4096,1024] @ [3072,1024]^T, scatter epilogue (q*scale, k, v^T)
    gemm_bf16_kernel<MODE_QKV><<<dim3(3 * C / 64, M / 128), 256, 0, stream>>>(
        h1, wqkvb, M, 3 * C, C, nullptr, nullptr, nullptr, nullptr, qs, kbuf, vT);

    // attention: 4096 waves, 8 per block
    attention_kernel<<<M / 8, 256, 0, stream>>>(qs, kbuf, vT, rpb, pmask, abuf);

    // proj + residual -> x2 (f32)
    gemm_bf16_kernel<MODE_PROJ><<<dim3(C / 64, M / 128), 256, 0, stream>>>(
        abuf, wprjb, M, C, C, b_proj, x, x2, nullptr, nullptr, nullptr, nullptr);

    // LN2
    layernorm_bf16_kernel<<<M, 256, 0, stream>>>(x2, ln2_g, ln2_b, h2, C);

    // FC1 + exact GELU -> u (bf16)
    gemm_bf16_kernel<MODE_GELU><<<dim3(HID / 64, M / 128), 256, 0, stream>>>(
        h2, wfc1b, M, HID, C, b_fc1, nullptr, nullptr, u, nullptr, nullptr, nullptr);

    // FC2 -> d_out[0..4M) = 2m, d_out[4M..8M) = m
    gemm_bf16_kernel<MODE_OUT2><<<dim3(C / 64, M / 128), 256, 0, stream>>>(
        u, wfc2b, M, C, HID, b_fc2, nullptr, out, nullptr, nullptr, nullptr, nullptr);

    (void)in_sizes; (void)n_in; (void)out_size; (void)ws_size;
}
